// PairwiseModalityAttention_15779709846342
// MI455X (gfx1250) — compile-verified, matrix operand order checked by reading, not run on hardware
//
#include <hip/hip_runtime.h>
#include <stdint.h>

// ---------------------------------------------------------------------------
// PairwiseModalityAttention for MI455X (gfx1250, wave32, WMMA).
//
// Math: softmax over a length-1 axis == 1.0 exactly, so Q/K projections are
// dead code. Each direction reduces to  mask_b * (LN_v(x_b) @ Wv'^T @ Wo^T)
// which we fold into a single fused weight  Weff = Wo @ (Wv * v_g)  and a
// per-row scalar epilogue (rsig, mu*rsig, mask). Per output modality the
// whole module is one B x 256 GEMM with K = D_self + D_k1 + D_k2 = 2304,
// done in bf16 WMMA (v_wmma_f32_16x16x32_bf16) with f32 accumulation,
// followed by the output LayerNorm.
//
// Tiles are staged with GLOBAL_LOAD_ASYNC_TO_LDS (ASYNCcnt) and a manual LDS
// double buffer: issue tile k+1 (5 async ops/thread), wait asynccnt<=5
// (in-order completion => tile k landed), barrier, WMMA.
// Falls back to load+ds_store staging if the async builtins are absent.
// ---------------------------------------------------------------------------

#define BQ   16384
#define HIDN 256
#define EPSV 1e-5f
#define ASTR 40   // LDS row stride in bf16 elems (32 + 8 pad; 80B rows, 16B aligned)

typedef __attribute__((ext_vector_type(16))) __bf16 v16bf;
typedef __attribute__((ext_vector_type(8)))  __bf16 v8bf;
typedef __attribute__((ext_vector_type(8)))  float  v8f;

#if defined(__HIP_DEVICE_COMPILE__) && \
    __has_builtin(__builtin_amdgcn_global_load_async_to_lds_b128) && \
    __has_builtin(__builtin_amdgcn_global_load_async_to_lds_b64)
#define USE_ASYNC_LDS 1
typedef int v2i_t __attribute__((ext_vector_type(2)));
typedef int v4i_t __attribute__((ext_vector_type(4)));
typedef v2i_t __attribute__((address_space(1)))* g_v2i_p;   // global src
typedef v4i_t __attribute__((address_space(1)))* g_v4i_p;
typedef v2i_t __attribute__((address_space(3)))* l_v2i_p;   // LDS dst
typedef v4i_t __attribute__((address_space(3)))* l_v4i_p;
#else
#define USE_ASYNC_LDS 0
#endif

static __device__ __forceinline__ unsigned short f2bf(float f) {
  unsigned int u = __float_as_uint(f);
  u += 0x7fffu + ((u >> 16) & 1u);          // round-to-nearest-even
  return (unsigned short)(u >> 16);
}
static __device__ __forceinline__ float bf2f(unsigned short h) {
  return __uint_as_float(((unsigned int)h) << 16);
}

// ---------------- per-row LN stats + f32 -> bf16 conversion ----------------
__global__ void stats_convert_kernel(const float* __restrict__ x,
                                     unsigned short* __restrict__ xbf,
                                     const int* __restrict__ mmask,
                                     float* __restrict__ rsig,
                                     float* __restrict__ tmus,
                                     float* __restrict__ maskf,
                                     int D, int midx) {
  __shared__ float s1[256], s2[256];
  const int row = blockIdx.x;
  const float* xr = x + (size_t)row * D;
  unsigned short* xo = xbf + (size_t)row * D;
  float a = 0.f, b = 0.f;
  for (int d = threadIdx.x; d < D; d += 256) {
    float v = xr[d];
    a += v; b += v * v;
    xo[d] = f2bf(v);
  }
  s1[threadIdx.x] = a; s2[threadIdx.x] = b;
  __syncthreads();
  for (int off = 128; off > 0; off >>= 1) {
    if ((int)threadIdx.x < off) {
      s1[threadIdx.x] += s1[threadIdx.x + off];
      s2[threadIdx.x] += s2[threadIdx.x + off];
    }
    __syncthreads();
  }
  if (threadIdx.x == 0) {
    float mu  = s1[0] / (float)D;
    float var = s2[0] / (float)D - mu * mu;
    float rs  = rsqrtf(var + EPSV);
    rsig[row]  = rs;
    tmus[row]  = mu * rs;
    maskf[row] = (mmask[row * 3 + midx] > 0) ? 1.0f : 0.0f;
  }
}

// --------------- bvp[j] = bv[j] + sum_d Wv[j][d] * v_b[d] ------------------
__global__ void bvp_kernel(const float* __restrict__ Wv, const float* __restrict__ bv,
                           const float* __restrict__ vb, float* __restrict__ bvp, int Dk) {
  int j = threadIdx.x;
  float acc = bv[j];
  const float* w = Wv + (size_t)j * Dk;
  for (int d = 0; d < Dk; ++d) acc += w[d] * vb[d];
  bvp[j] = acc;
}

// --------------------- c[n] = sum_j Wo[n][j] * bvp[j] ----------------------
__global__ void cvec_kernel(const float* __restrict__ Wo, const float* __restrict__ bvp,
                            float* __restrict__ cv) {
  int n = threadIdx.x;
  float acc = 0.f;
  const float* w = Wo + (size_t)n * HIDN;
  for (int j = 0; j < HIDN; ++j) acc += w[j] * bvp[j];
  cv[n] = acc;
}

// ----- Weff[n][k] = (sum_j Wo[n][j]*Wv[j][k]) * v_g[k], stored as bf16 -----
__global__ void weff_kernel(const float* __restrict__ Wo, const float* __restrict__ Wv,
                            const float* __restrict__ vg,
                            unsigned short* __restrict__ weff, int Dk) {
  __shared__ float wo_s[16 * 256];
  const int k = blockIdx.x * 256 + threadIdx.x;
  const int nbase = blockIdx.y * 16;
  for (int i = threadIdx.x; i < 16 * 256; i += 256)
    wo_s[i] = Wo[(size_t)(nbase + (i >> 8)) * HIDN + (i & 255)];
  __syncthreads();
  float acc[16];
#pragma unroll
  for (int n = 0; n < 16; ++n) acc[n] = 0.f;
  for (int j = 0; j < HIDN; ++j) {
    float wv = Wv[(size_t)j * Dk + k];
#pragma unroll
    for (int n = 0; n < 16; ++n) acc[n] += wo_s[n * 256 + j] * wv;
  }
  const float g = vg[k];
#pragma unroll
  for (int n = 0; n < 16; ++n)
    weff[(size_t)(nbase + n) * Dk + k] = f2bf(acc[n] * g);
}

// ---------------- s[n] = sum_k Weff_bf16[n][k] (deterministic) -------------
__global__ void svec_kernel(const unsigned short* __restrict__ weff,
                            float* __restrict__ svec, int Dk) {
  int n = threadIdx.x;
  const unsigned short* w = weff + (size_t)n * Dk;
  float acc = 0.f;
  for (int k = 0; k < Dk; ++k) acc += bf2f(w[k]);
  svec[n] = acc;
}

// ------------------------- f32 -> bf16 elementwise -------------------------
__global__ void conv_bf16_kernel(const float* __restrict__ in,
                                 unsigned short* __restrict__ out, int n) {
  int i = blockIdx.x * 256 + threadIdx.x;
  if (i < n) out[i] = f2bf(in[i]);
}

// ---------------- cbias[h] = b_self[h] + bo_dir1[h] + bo_dir2[h] -----------
__global__ void cbias_kernel(const float* __restrict__ bself, const float* __restrict__ bo1,
                             const float* __restrict__ bo2, float* __restrict__ cb) {
  int h = threadIdx.x;
  cb[h] = bself[h] + bo1[h] + bo2[h];
}

// ------------- stage one 32-row x 32-k A tile + 256 x 32-k B tile ----------
// Async path: 1x b64 (A) + 4x b128 (B) per thread = 5 ASYNCcnt ops.
static __device__ __forceinline__ void issue_tile(
    const unsigned short* __restrict__ X, const unsigned short* __restrict__ W,
    int D, int blockRow, int k0, int tid,
    unsigned short* As, unsigned short* Bs) {
  const int idx = tid * 4;
  const int r = idx >> 5, kk = idx & 31;
  const unsigned short* asrc = X + (size_t)(blockRow + r) * D + k0 + kk;
  const unsigned short* bsrc = W + (size_t)tid * D + k0;
  unsigned short* adst = As + r * ASTR + kk;
  unsigned short* bdst = Bs + tid * ASTR;
#if USE_ASYNC_LDS
  __builtin_amdgcn_global_load_async_to_lds_b64((g_v2i_p)asrc, (l_v2i_p)adst, 0, 0);
  __builtin_amdgcn_global_load_async_to_lds_b128((g_v4i_p)(bsrc + 0),  (l_v4i_p)(bdst + 0),  0, 0);
  __builtin_amdgcn_global_load_async_to_lds_b128((g_v4i_p)(bsrc + 8),  (l_v4i_p)(bdst + 8),  0, 0);
  __builtin_amdgcn_global_load_async_to_lds_b128((g_v4i_p)(bsrc + 16), (l_v4i_p)(bdst + 16), 0, 0);
  __builtin_amdgcn_global_load_async_to_lds_b128((g_v4i_p)(bsrc + 24), (l_v4i_p)(bdst + 24), 0, 0);
#else
  *(uint2*)adst = *(const uint2*)asrc;
  uint4 d0 = *(const uint4*)(bsrc);
  uint4 d1 = *(const uint4*)(bsrc + 8);
  uint4 d2 = *(const uint4*)(bsrc + 16);
  uint4 d3 = *(const uint4*)(bsrc + 24);
  *(uint4*)(bdst + 0)  = d0;
  *(uint4*)(bdst + 8)  = d1;
  *(uint4*)(bdst + 16) = d2;
  *(uint4*)(bdst + 24) = d3;
#endif
}

// prefetched==true: allow the 5 just-issued next-tile ops to remain in flight.
static __device__ __forceinline__ void wait_tile(bool prefetched) {
#if USE_ASYNC_LDS
#if __has_builtin(__builtin_amdgcn_s_wait_asynccnt)
  if (prefetched) __builtin_amdgcn_s_wait_asynccnt(5);
  else            __builtin_amdgcn_s_wait_asynccnt(0);
#else
  if (prefetched) asm volatile("s_wait_asynccnt 0x5" ::: "memory");
  else            asm volatile("s_wait_asynccnt 0x0" ::: "memory");
#endif
#else
  (void)prefetched;
#endif
}

// ---------------------------------------------------------------------------
// Fused GEMM: out_pre[b,h] = (x_self @ Ws^T)
//   + m1*rsig1*(x1 @ W1^T) + m1*(c1[h] - mu1*rsig1*s1[h])
//   + m2*rsig2*(x2 @ W2^T) + m2*(c2[h] - mu2*rsig2*s2[h]) + cbias[h]
// Block: 256 threads = 8 waves; tile 32 rows x 256 cols, LDS double buffer.
// Wave (rg,cg): rows 16*rg.., cols 64*cg.. ; 3 segments x 4 col-tiles accums.
// A/B WMMA fragments (bf16 16x16x32): lane L holds row/col = L%16,
// K elems {o..o+7, 16+o..16+o+7}, o = 8*(L>=16).
// ---------------------------------------------------------------------------
__global__ __launch_bounds__(256) void fused_gemm_kernel(
    const unsigned short* __restrict__ xs, int Ds, const unsigned short* __restrict__ Wsw,
    const unsigned short* __restrict__ x1, int D1, const unsigned short* __restrict__ W1,
    const unsigned short* __restrict__ x2, int D2, const unsigned short* __restrict__ W2,
    const float* __restrict__ rsig1, const float* __restrict__ t1, const float* __restrict__ mk1,
    const float* __restrict__ rsig2, const float* __restrict__ t2, const float* __restrict__ mk2,
    const float* __restrict__ s1v, const float* __restrict__ c1v,
    const float* __restrict__ s2v, const float* __restrict__ c2v,
    const float* __restrict__ cbias, float* __restrict__ outpre) {
  __shared__ unsigned short As[2][32 * ASTR];
  __shared__ unsigned short Bs[2][256 * ASTR];
  const int tid  = threadIdx.x;
  const int lane = tid & 31;
  const int wave = tid >> 5;
  const int rowGroup = wave >> 2;   // 0..1
  const int colGroup = wave & 3;    // 0..3
  const int blockRow = blockIdx.x * 32;
  const int hl  = lane >> 4;        // half of wave
  const int l16 = lane & 15;
  const int ko  = hl * 8;           // K sub-offset for this half

  v8f acc[3][4];
  const v8f vzero = {0.f, 0.f, 0.f, 0.f, 0.f, 0.f, 0.f, 0.f};
#pragma unroll
  for (int s = 0; s < 3; ++s)
#pragma unroll
    for (int c = 0; c < 4; ++c) acc[s][c] = vzero;

  const unsigned short* const xPtr[3] = {xs, x1, x2};
  const unsigned short* const wPtr[3] = {Wsw, W1, W2};
  const int Dseg[3] = {Ds, D1, D2};

  union U { v8bf v; uint4 u; };

#pragma unroll
  for (int seg = 0; seg < 3; ++seg) {
    const unsigned short* X = xPtr[seg];
    const unsigned short* W = wPtr[seg];
    const int D = Dseg[seg];
    const int steps = D >> 5;
    issue_tile(X, W, D, blockRow, 0, tid, As[0], Bs[0]);
    for (int kstep = 0; kstep < steps; ++kstep) {
      const int cur = kstep & 1, nxt = cur ^ 1;
      const bool pf = (kstep + 1 < steps);
      if (pf) issue_tile(X, W, D, blockRow, (kstep + 1) * 32, tid, As[nxt], Bs[nxt]);
      wait_tile(pf);
      __syncthreads();
      U alo, ahi;
      alo.u = *(const uint4*)&As[cur][(rowGroup * 16 + l16) * ASTR + ko];
      ahi.u = *(const uint4*)&As[cur][(rowGroup * 16 + l16) * ASTR + 16 + ko];
      v16bf afrag = __builtin_shufflevector(alo.v, ahi.v,
          0, 1, 2, 3, 4, 5, 6, 7, 8, 9, 10, 11, 12, 13, 14, 15);
#pragma unroll
      for (int ct = 0; ct < 4; ++ct) {
        int n = colGroup * 64 + ct * 16 + l16;
        U blo, bhi;
        blo.u = *(const uint4*)&Bs[cur][n * ASTR + ko];
        bhi.u = *(const uint4*)&Bs[cur][n * ASTR + 16 + ko];
        v16bf bfrag = __builtin_shufflevector(blo.v, bhi.v,
            0, 1, 2, 3, 4, 5, 6, 7, 8, 9, 10, 11, 12, 13, 14, 15);
        acc[seg][ct] = __builtin_amdgcn_wmma_f32_16x16x32_bf16(
            false, afrag, false, bfrag, (short)0, acc[seg][ct], false, false);
      }
      __syncthreads();
    }
  }

  // Epilogue. D layout: lane l16 -> col, VGPR r -> row (r + 8*hl).
  float s1h[4], c1h[4], s2h[4], c2h[4], cbh[4];
#pragma unroll
  for (int ct = 0; ct < 4; ++ct) {
    int h = colGroup * 64 + ct * 16 + l16;
    s1h[ct] = s1v[h]; c1h[ct] = c1v[h];
    s2h[ct] = s2v[h]; c2h[ct] = c2v[h];
    cbh[ct] = cbias[h];
  }
#pragma unroll
  for (int r = 0; r < 8; ++r) {
    int row = blockRow + rowGroup * 16 + hl * 8 + r;
    float m1 = mk1[row], a1 = m1 * rsig1[row], tv1 = t1[row];
    float m2 = mk2[row], a2 = m2 * rsig2[row], tv2 = t2[row];
#pragma unroll
    for (int ct = 0; ct < 4; ++ct) {
      int h = colGroup * 64 + ct * 16 + l16;
      float pre = acc[0][ct][r]
                + a1 * acc[1][ct][r] + m1 * (c1h[ct] - tv1 * s1h[ct])
                + a2 * acc[2][ct][r] + m2 * (c2h[ct] - tv2 * s2h[ct])
                + cbh[ct];
      outpre[(size_t)row * HIDN + h] = pre;
    }
  }
}

// ----------------------- output LayerNorm + avail mask ---------------------
__global__ void out_ln_kernel(const float* __restrict__ pre,
                              const float* __restrict__ g, const float* __restrict__ bln,
                              const float* __restrict__ maskf, float* __restrict__ out) {
  __shared__ float s1[256], s2[256];
  const int row = blockIdx.x;
  float v = pre[(size_t)row * HIDN + threadIdx.x];
  s1[threadIdx.x] = v; s2[threadIdx.x] = v * v;
  __syncthreads();
  for (int off = 128; off > 0; off >>= 1) {
    if ((int)threadIdx.x < off) {
      s1[threadIdx.x] += s1[threadIdx.x + off];
      s2[threadIdx.x] += s2[threadIdx.x + off];
    }
    __syncthreads();
  }
  float mu  = s1[0] / (float)HIDN;
  float var = s2[0] / (float)HIDN - mu * mu;
  float rs  = rsqrtf(var + EPSV);
  out[(size_t)row * HIDN + threadIdx.x] =
      ((v - mu) * rs * g[threadIdx.x] + bln[threadIdx.x]) * maskf[row];
}

// ----------- attn maps: softmax over Tk=1 is exactly 1, masked to 0 --------
__global__ void attn_kernel(const int* __restrict__ mmask, float* __restrict__ out) {
  int i = blockIdx.x * 256 + threadIdx.x;   // over 6*B*4
  if (i >= 6 * BQ * 4) return;
  int b   = (i >> 2) % BQ;
  int dir = i / (BQ * 4);
  const int keyIdx[6] = {1, 0, 2, 0, 2, 1};   // DIRS key modality
  out[i] = (mmask[b * 3 + keyIdx[dir]] > 0) ? 1.0f : 0.0f;
}

// ---------------------------------------------------------------------------
extern "C" void kernel_launch(void* const* d_in, const int* in_sizes, int n_in,
                              void* d_out, int out_size, void* d_ws, size_t ws_size,
                              hipStream_t stream) {
  (void)n_in; (void)out_size; (void)ws_size;
  static const int Dm[3]     = {512, 768, 1024};       // audio, video, text
  static const int keyMod[6] = {1, 0, 2, 0, 2, 1};     // DIRS key modality
  static const int Dk[6]     = {768, 512, 1024, 512, 1024, 768};
  static const int dirOf[3][2] = {{0, 2}, {1, 4}, {3, 5}};  // dirs feeding msgs[m]

  const float* x[3] = {(const float*)d_in[0], (const float*)d_in[1], (const float*)d_in[2]};
  const int* mmask  = (const int*)d_in[3];

  // -- locate param leaves: jax pytree sorted-key order vs dict insertion --
  const bool srt = (in_sizes[4] == 256 * 1024);
  int dirBase[6], oWv, oBv, oWo, oBo, oVg, oVb, selfW[3], selfB[3], lnG, lnB;
  if (srt) {
    const int bb[6] = {18, 68, 4, 40, 82, 54};  // a2v, v2a, a2t, t2a, v2t, t2v
    for (int i = 0; i < 6; ++i) dirBase[i] = bb[i];
    oWv = 3; oBv = 7; oWo = 1; oBo = 5; oVg = 13; oVb = 12;
    selfW[0] = 34; selfB[0] = 35; selfW[1] = 38; selfB[1] = 39; selfW[2] = 36; selfB[2] = 37;
    lnG = 33; lnB = 32;
  } else {
    for (int i = 0; i < 6; ++i) dirBase[i] = 4 + 14 * i;
    oWv = 10; oBv = 11; oWo = 12; oBo = 13; oVg = 4; oVb = 5;
    selfW[0] = 88; selfB[0] = 89; selfW[1] = 90; selfB[1] = 91; selfW[2] = 92; selfB[2] = 93;
    lnG = 94; lnB = 95;
  }
  const float *pWv[6], *pBv[6], *pWo[6], *pBo[6], *pVg[6], *pVb[6];
  for (int i = 0; i < 6; ++i) {
    pWv[i] = (const float*)d_in[dirBase[i] + oWv];
    pBv[i] = (const float*)d_in[dirBase[i] + oBv];
    pWo[i] = (const float*)d_in[dirBase[i] + oWo];
    pBo[i] = (const float*)d_in[dirBase[i] + oBo];
    pVg[i] = (const float*)d_in[dirBase[i] + oVg];
    pVb[i] = (const float*)d_in[dirBase[i] + oVb];
  }
  const float* pSelfW[3]; const float* pSelfB[3];
  for (int m = 0; m < 3; ++m) {
    pSelfW[m] = (const float*)d_in[selfW[m]];
    pSelfB[m] = (const float*)d_in[selfB[m]];
  }
  const float* outLnG = (const float*)d_in[lnG];
  const float* outLnB = (const float*)d_in[lnB];

  // ---------------------------- workspace layout ---------------------------
  char* wsb = (char*)d_ws;
  size_t cur = 0;
  auto alloc = [&](size_t bytes) -> char* {
    char* p = wsb + cur;
    cur = (cur + bytes + 255) & ~(size_t)255;
    return p;
  };
  unsigned short* xbf[3]; float *rsig[3], *tmus[3], *maskf[3];
  for (int m = 0; m < 3; ++m) {
    xbf[m]   = (unsigned short*)alloc((size_t)BQ * Dm[m] * 2);
    rsig[m]  = (float*)alloc((size_t)BQ * 4);
    tmus[m]  = (float*)alloc((size_t)BQ * 4);
    maskf[m] = (float*)alloc((size_t)BQ * 4);
  }
  unsigned short* weff[6]; float *svec[6], *cvec[6], *bvp[6];
  for (int i = 0; i < 6; ++i) {
    weff[i] = (unsigned short*)alloc((size_t)HIDN * Dk[i] * 2);
    svec[i] = (float*)alloc(HIDN * 4);
    cvec[i] = (float*)alloc(HIDN * 4);
    bvp[i]  = (float*)alloc(HIDN * 4);
  }
  unsigned short* wselfbf[3]; float* cbias[3]; float* pre[3];
  for (int m = 0; m < 3; ++m) {
    wselfbf[m] = (unsigned short*)alloc((size_t)HIDN * Dm[m] * 2);
    cbias[m]   = (float*)alloc(HIDN * 4);
    pre[m]     = (float*)alloc((size_t)BQ * HIDN * 4);
  }

  // ------------------------------- pipeline --------------------------------
  for (int m = 0; m < 3; ++m)
    stats_convert_kernel<<<BQ, 256, 0, stream>>>(x[m], xbf[m], mmask,
                                                 rsig[m], tmus[m], maskf[m], Dm[m], m);
  for (int i = 0; i < 6; ++i) {
    bvp_kernel<<<1, 256, 0, stream>>>(pWv[i], pBv[i], pVb[i], bvp[i], Dk[i]);
    cvec_kernel<<<1, 256, 0, stream>>>(pWo[i], bvp[i], cvec[i]);
    dim3 g(Dk[i] / 256, 16);
    weff_kernel<<<g, 256, 0, stream>>>(pWo[i], pWv[i], pVg[i], weff[i], Dk[i]);
    svec_kernel<<<1, 256, 0, stream>>>(weff[i], svec[i], Dk[i]);
  }
  for (int m = 0; m < 3; ++m) {
    int n = HIDN * Dm[m];
    conv_bf16_kernel<<<(n + 255) / 256, 256, 0, stream>>>(pSelfW[m], wselfbf[m], n);
    int d1 = dirOf[m][0], d2 = dirOf[m][1];
    cbias_kernel<<<1, 256, 0, stream>>>(pSelfB[m], pBo[d1], pBo[d2], cbias[m]);
  }
  for (int m = 0; m < 3; ++m) {
    int d1 = dirOf[m][0], d2 = dirOf[m][1];
    int k1 = keyMod[d1], k2 = keyMod[d2];
    fused_gemm_kernel<<<BQ / 32, 256, 0, stream>>>(
        xbf[m], Dm[m], wselfbf[m],
        xbf[k1], Dm[k1], weff[d1],
        xbf[k2], Dm[k2], weff[d2],
        rsig[k1], tmus[k1], maskf[k1],
        rsig[k2], tmus[k2], maskf[k2],
        svec[d1], cvec[d1], svec[d2], cvec[d2],
        cbias[m], pre[m]);
  }
  float* out = (float*)d_out;
  for (int m = 0; m < 3; ++m)
    out_ln_kernel<<<BQ, 256, 0, stream>>>(pre[m], outLnG, outLnB, maskf[m],
                                          out + (size_t)m * BQ * HIDN);
  attn_kernel<<<(6 * BQ * 4) / 256, 256, 0, stream>>>(mmask, out + (size_t)3 * BQ * HIDN);
}
